// SparseMSDeformableAttentionBlock_2052994367932
// MI455X (gfx1250) — matrix operand order repaired and checked by reading, not compiled, not run on hardware
//
#include <hip/hip_runtime.h>

#define EMBED 256
#define HEADS 8
#define LEVELS 4
#define POINTS 4
#define HEAD_DIM 32
#define BATCH 2
#define NQ 20000
#define HPIX 128
#define WPIX 128
#define VROWS (BATCH*HPIX*WPIX*LEVELS)   // 131072

typedef __attribute__((ext_vector_type(16))) __bf16 v16bf;
typedef __attribute__((ext_vector_type(8)))  __bf16 v8bf;
typedef __attribute__((ext_vector_type(2)))  __bf16 v2bf;
typedef __attribute__((ext_vector_type(8)))  float  v8f;
typedef __attribute__((ext_vector_type(4)))  float  v4f;

static __device__ __forceinline__ v8f wmma_bf16(v16bf a, v16bf b, v8f c) {
  // 8 args: (neg_a, A, neg_b, B, c_mod, C, reuse_a, reuse_b)
  return __builtin_amdgcn_wmma_f32_16x16x32_bf16(false, a, false, b, (short)0, c, false, false);
}

// B fragment from pre-packed weight buffer: layout [nt][k0][lane][16e], 32B per lane.
static __device__ __forceinline__ v16bf ldb(const __bf16* __restrict__ Wf, int nt, int k0,
                                            int lane, int kdim) {
  return *(const v16bf*)(Wf + ((((nt * (kdim >> 5)) + k0) * 32 + lane) << 4));
}

// A fragment from row-major bf16: base already points at (row, k0*32 + khalf*8).
// elements 0..7 = 8 contiguous bf16, elements 8..15 = 8 contiguous bf16 at +16 cols.
static __device__ __forceinline__ v16bf lda_rm(const __bf16* __restrict__ base) {
  v8bf lo = *(const v8bf*)base;
  v8bf hi = *(const v8bf*)(base + 16);
  return __builtin_shufflevector(lo, hi, 0,1,2,3,4,5,6,7,8,9,10,11,12,13,14,15);
}

// Build one A fragment from a row-major fp32 row (converts to bf16).
static __device__ __forceinline__ v16bf lda_f32(const float* __restrict__ row, int k0, int khalf) {
  v16bf a;
  const int c0 = k0 * 32 + khalf * 8;
  #pragma unroll
  for (int half = 0; half < 2; ++half) {
    const int c = c0 + half * 16;
    v4f x0 = *(const v4f*)(row + c);
    v4f x1 = *(const v4f*)(row + c + 4);
    #pragma unroll
    for (int j = 0; j < 8; ++j)
      a[half * 8 + j] = (__bf16)(j < 4 ? x0[j] : x1[j - 4]);
  }
  return a;
}

// ---------------------------------------------------------------------------
// Kernel 0: repack fp32 weight [K][N] into bf16 B-fragment layout.
// B 32x16 bf16 layout: n = nt*16 + (lane&15); K = k0*32 + 8*(lane>>4) + (e&7) + 16*(e>>3)
// ---------------------------------------------------------------------------
__global__ void k_wconv(const float* __restrict__ src, __bf16* __restrict__ dst,
                        int kdim, int ndim) {
  int idx = blockIdx.x * 256 + threadIdx.x;
  if (idx >= kdim * ndim) return;
  int e    = idx & 15;
  int lane = (idx >> 4) & 31;
  int t    = idx >> 9;
  int kt   = kdim >> 5;
  int k0   = t % kt;
  int nt   = t / kt;
  int n    = nt * 16 + (lane & 15);
  int kk   = k0 * 32 + ((lane >> 4) << 3) + (e & 7) + ((e >> 3) << 4);
  dst[idx] = (__bf16)src[(size_t)kk * ndim + n];
}

// ---------------------------------------------------------------------------
// Kernel 1: per-16-query tile: LayerNorm -> x@W_q -> offsets(->pix) + attn softmax
// one wave per tile, 4 waves / block; per-wave LDS holds xq in bf16.
// ---------------------------------------------------------------------------
__global__ void __launch_bounds__(128) k_query(
    const float* __restrict__ query, const float* __restrict__ qpos,
    const float* __restrict__ gamma, const float* __restrict__ beta,
    const __bf16* __restrict__ Wqf, const __bf16* __restrict__ Woff,
    const __bf16* __restrict__ Watf,
    const float* __restrict__ b_off, const float* __restrict__ b_attn,
    const int* __restrict__ shapes,
    float* __restrict__ pix_out, float* __restrict__ attn_out) {
  __shared__ __bf16 lds_xq[4][16 * 264];   // stride 264 bf16 -> conflict-free ds_read_b128
  const int wave  = threadIdx.x >> 5;
  const int lane  = threadIdx.x & 31;
  const int tile  = blockIdx.x * 4 + wave;
  if (tile >= NQ / 16) return;
  const int qbase = tile * 16;
  const int mrow  = lane & 15;
  const int khalf = lane >> 4;
  const float* qrow = query + (size_t)(qbase + mrow) * EMBED;

  // ---- LayerNorm pass 1: this lane covers half the row (128 elems) ----
  float s = 0.f, s2 = 0.f;
  #pragma unroll
  for (int k0 = 0; k0 < 8; ++k0) {
    const int c0 = k0 * 32 + khalf * 8;
    #pragma unroll
    for (int half = 0; half < 2; ++half) {
      const float* p = qrow + c0 + half * 16;
      v4f x0 = *(const v4f*)p;
      v4f x1 = *(const v4f*)(p + 4);
      #pragma unroll
      for (int j = 0; j < 4; ++j) {
        s  += x0[j] + x1[j];
        s2 += x0[j] * x0[j] + x1[j] * x1[j];
      }
    }
  }
  s  += __shfl_xor(s, 16, 32);
  s2 += __shfl_xor(s2, 16, 32);
  const float mu   = s * (1.f / EMBED);
  const float var  = s2 * (1.f / EMBED) - mu * mu;
  const float rstd = rsqrtf(var + 1e-5f);

  // ---- pass 2: reload (cache-hot), normalize, build bf16 A fragments in regs ----
  v16bf a[8];
  #pragma unroll
  for (int k0 = 0; k0 < 8; ++k0) {
    const int c0 = k0 * 32 + khalf * 8;
    #pragma unroll
    for (int half = 0; half < 2; ++half) {
      const int c = c0 + half * 16;
      v4f x0 = *(const v4f*)(qrow + c);
      v4f x1 = *(const v4f*)(qrow + c + 4);
      #pragma unroll
      for (int j = 0; j < 8; ++j) {
        float xv = ((j < 4 ? x0[j] : x1[j - 4]) - mu) * rstd * gamma[c + j] + beta[c + j];
        a[k0][half * 8 + j] = (__bf16)xv;
      }
    }
  }

  // ---- xq = LN(x) @ W_q, stored bf16 into per-wave LDS ----
  __bf16* xq = lds_xq[wave];
  for (int nt = 0; nt < 16; ++nt) {
    v8f acc = {0, 0, 0, 0, 0, 0, 0, 0};
    #pragma unroll
    for (int k0 = 0; k0 < 8; ++k0)
      acc = wmma_bf16(a[k0], ldb(Wqf, nt, k0, lane, EMBED), acc);
    #pragma unroll
    for (int r = 0; r < 8; ++r)
      xq[(khalf * 8 + r) * 264 + nt * 16 + mrow] = (__bf16)acc[r];
  }
  asm volatile("s_wait_dscnt 0" ::: "memory");

  // ---- off = xq @ W_off + b_off  -> fused into pixel coords ----
  // pix = qsp[c]*shp[l][c] + off - 0.5 ; n = h*32 + l*8 + p*2 + c
  for (int nt = 0; nt < 16; ++nt) {
    v8f acc = {0, 0, 0, 0, 0, 0, 0, 0};
    #pragma unroll
    for (int k0 = 0; k0 < 8; ++k0)
      acc = wmma_bf16(lda_rm(xq + mrow * 264 + k0 * 32 + khalf * 8),
                      ldb(Woff, nt, k0, lane, EMBED), acc);
    const int n = nt * 16 + mrow;
    const float bo = b_off[n];
    const int c = n & 1;
    const int l = (n >> 3) & 3;
    const float shp = (float)shapes[l * 2 + c];
    #pragma unroll
    for (int r = 0; r < 8; ++r) {
      const int q = qbase + khalf * 8 + r;
      pix_out[(size_t)q * 256 + n] = qpos[q * 2 + c] * shp + (acc[r] + bo) - 0.5f;
    }
  }

  // ---- attn = softmax over 16 (l,p) per head; one head == one N-tile ----
  for (int nt = 0; nt < 8; ++nt) {
    v8f acc = {0, 0, 0, 0, 0, 0, 0, 0};
    #pragma unroll
    for (int k0 = 0; k0 < 8; ++k0)
      acc = wmma_bf16(lda_rm(xq + mrow * 264 + k0 * 32 + khalf * 8),
                      ldb(Watf, nt, k0, lane, EMBED), acc);
    const int n = nt * 16 + mrow;
    const float ba = b_attn[n];
    #pragma unroll
    for (int r = 0; r < 8; ++r) {
      float v = acc[r] + ba;
      float m = v;
      #pragma unroll
      for (int msk = 8; msk >= 1; msk >>= 1) m = fmaxf(m, __shfl_xor(m, msk, 32));
      float e = __expf(v - m);
      float ss = e;
      #pragma unroll
      for (int msk = 8; msk >= 1; msk >>= 1) ss += __shfl_xor(ss, msk, 32);
      const int q = qbase + khalf * 8 + r;
      attn_out[(size_t)q * 128 + n] = e / ss;
    }
  }
}

// ---------------------------------------------------------------------------
// Kernel 2: value = fmaps @ W_val + b_val, bf16 out.
// 32 rows (2 A tiles) per wave: each B fragment loaded once, used twice.
// ---------------------------------------------------------------------------
__global__ void __launch_bounds__(256) k_value(
    const float* __restrict__ fmaps, const __bf16* __restrict__ Wvf,
    const float* __restrict__ b_val, __bf16* __restrict__ vout) {
  const int wave  = threadIdx.x >> 5;
  const int lane  = threadIdx.x & 31;
  const int job   = blockIdx.x * 8 + wave;
  const int rbase = job * 32;
  const int mrow  = lane & 15;
  const int khalf = lane >> 4;
  const float* arow0 = fmaps + (size_t)(rbase + mrow) * EMBED;
  const float* arow1 = fmaps + (size_t)(rbase + 16 + mrow) * EMBED;

  v16bf a0[8], a1[8];
  #pragma unroll
  for (int k0 = 0; k0 < 8; ++k0) {
    a0[k0] = lda_f32(arow0, k0, khalf);
    a1[k0] = lda_f32(arow1, k0, khalf);
  }

  for (int nt = 0; nt < 16; ++nt) {
    v8f acc0 = {0, 0, 0, 0, 0, 0, 0, 0};
    v8f acc1 = {0, 0, 0, 0, 0, 0, 0, 0};
    #pragma unroll
    for (int k0 = 0; k0 < 8; ++k0) {
      v16bf b = ldb(Wvf, nt, k0, lane, EMBED);
      acc0 = wmma_bf16(a0[k0], b, acc0);
      acc1 = wmma_bf16(a1[k0], b, acc1);
    }
    const int n = nt * 16 + mrow;
    const float bv = b_val[n];
    #pragma unroll
    for (int r = 0; r < 8; ++r) {
      vout[(size_t)(rbase + khalf * 8 + r) * EMBED + n]      = (__bf16)(acc0[r] + bv);
      vout[(size_t)(rbase + 16 + khalf * 8 + r) * EMBED + n] = (__bf16)(acc1[r] + bv);
    }
  }
}

// ---------------------------------------------------------------------------
// Kernel 3: bilinear sampling + attention-weighted reduction over (l,p).
// One query per wave; lane = (head-half, dim-pair) => 64B coalesced corner reads.
// ---------------------------------------------------------------------------
__global__ void __launch_bounds__(256) k_sample(
    const float* __restrict__ pix_in, const float* __restrict__ attnw,
    const __bf16* __restrict__ val, const int* __restrict__ qbo,
    const int* __restrict__ shapes, __bf16* __restrict__ samp) {
  const int wave = threadIdx.x >> 5;
  const int lane = threadIdx.x & 31;
  const int q = blockIdx.x * 8 + wave;
  if (q >= NQ) return;
  int b = 0;
  #pragma unroll
  for (int i = 1; i < BATCH; ++i)
    if (q >= qbo[i]) b = i;
  const int d0 = (lane & 15) * 2;

  #pragma unroll
  for (int hh = 0; hh < 4; ++hh) {
    const int h = hh * 2 + (lane >> 4);
    float a0 = 0.f, a1 = 0.f;
    for (int l = 0; l < LEVELS; ++l) {
      const int Hl = shapes[l * 2];
      const int Wl = shapes[l * 2 + 1];
      #pragma unroll
      for (int p = 0; p < POINTS; ++p) {
        const int n = h * 32 + l * 8 + p * 2;
        const float py = pix_in[(size_t)q * 256 + n];
        const float px = pix_in[(size_t)q * 256 + n + 1];
        const float aw = attnw[(size_t)q * 128 + h * 16 + l * 4 + p];
        const float y0f = floorf(py), x0f = floorf(px);
        const float wy = py - y0f, wx = px - x0f;
        const int y0 = (int)y0f, x0 = (int)x0f;
        #pragma unroll
        for (int cr = 0; cr < 4; ++cr) {
          const int dy = cr >> 1, dx = cr & 1;
          const int yi = y0 + dy, xi = x0 + dx;
          const float cw = (dy ? wy : 1.f - wy) * (dx ? wx : 1.f - wx);
          const float w = ((yi >= 0) && (yi < Hl) && (xi >= 0) && (xi < Wl)) ? cw * aw : 0.f;
          const int yc = min(max(yi, 0), HPIX - 1);
          const int xc = min(max(xi, 0), WPIX - 1);
          const __bf16* vp = val +
              ((size_t)(((b * HPIX + yc) * WPIX + xc) * LEVELS + l)) * EMBED +
              h * HEAD_DIM + d0;
          v2bf vv = *(const v2bf*)vp;
          a0 += w * (float)vv[0];
          a1 += w * (float)vv[1];
        }
      }
    }
    v2bf o;
    o[0] = (__bf16)a0;
    o[1] = (__bf16)a1;
    *(v2bf*)(samp + (size_t)q * 256 + h * HEAD_DIM + d0) = o;
  }
}

// ---------------------------------------------------------------------------
// Kernel 4: out = sampled @ W_out + residual(query)
// 32 rows (2 A tiles) per wave, B fragment reused across both.
// ---------------------------------------------------------------------------
__global__ void __launch_bounds__(256) k_out(
    const __bf16* __restrict__ samp, const __bf16* __restrict__ Wof,
    const float* __restrict__ query, float* __restrict__ out) {
  const int wave  = threadIdx.x >> 5;
  const int lane  = threadIdx.x & 31;
  const int job   = blockIdx.x * 8 + wave;
  if (job >= NQ / 32) return;          // 625 wave-jobs
  const int qbase = job * 32;
  const int mrow  = lane & 15;
  const int khalf = lane >> 4;
  const __bf16* arow0 = samp + (size_t)(qbase + mrow) * EMBED;
  const __bf16* arow1 = samp + (size_t)(qbase + 16 + mrow) * EMBED;

  v16bf a0[8], a1[8];
  #pragma unroll
  for (int k0 = 0; k0 < 8; ++k0) {
    a0[k0] = lda_rm(arow0 + k0 * 32 + khalf * 8);
    a1[k0] = lda_rm(arow1 + k0 * 32 + khalf * 8);
  }

  for (int nt = 0; nt < 16; ++nt) {
    v8f acc0 = {0, 0, 0, 0, 0, 0, 0, 0};
    v8f acc1 = {0, 0, 0, 0, 0, 0, 0, 0};
    #pragma unroll
    for (int k0 = 0; k0 < 8; ++k0) {
      v16bf b = ldb(Wof, nt, k0, lane, EMBED);
      acc0 = wmma_bf16(a0[k0], b, acc0);
      acc1 = wmma_bf16(a1[k0], b, acc1);
    }
    const int n = nt * 16 + mrow;
    #pragma unroll
    for (int r = 0; r < 8; ++r) {
      const size_t qi0 = (size_t)(qbase + khalf * 8 + r) * EMBED + n;
      const size_t qi1 = (size_t)(qbase + 16 + khalf * 8 + r) * EMBED + n;
      out[qi0] = acc0[r] + query[qi0];
      out[qi1] = acc1[r] + query[qi1];
    }
  }
}

// ---------------------------------------------------------------------------
extern "C" void kernel_launch(void* const* d_in, const int* in_sizes, int n_in,
                              void* d_out, int out_size, void* d_ws, size_t ws_size,
                              hipStream_t stream) {
  (void)in_sizes; (void)n_in; (void)out_size; (void)ws_size;
  const float* query  = (const float*)d_in[0];
  const float* qpos   = (const float*)d_in[1];
  const int*   qbo    = (const int*)d_in[2];
  const float* fmaps  = (const float*)d_in[3];
  const int*   shapes = (const int*)d_in[4];
  const float* gamma  = (const float*)d_in[5];
  const float* beta   = (const float*)d_in[6];
  const float* W_q    = (const float*)d_in[7];
  const float* W_off  = (const float*)d_in[8];
  const float* b_off  = (const float*)d_in[9];
  const float* W_attn = (const float*)d_in[10];
  const float* b_attn = (const float*)d_in[11];
  const float* W_val  = (const float*)d_in[12];
  const float* b_val  = (const float*)d_in[13];
  const float* W_out  = (const float*)d_in[14];
  float* out = (float*)d_out;
  char*  ws  = (char*)d_ws;

  // workspace layout (bytes)
  __bf16* Wqf  = (__bf16*)(ws + 0);          // 128K
  __bf16* Woff = (__bf16*)(ws + 131072);     // 128K
  __bf16* Watf = (__bf16*)(ws + 262144);     // 64K
  __bf16* Wvf  = (__bf16*)(ws + 327680);     // 128K
  __bf16* Wof  = (__bf16*)(ws + 458752);     // 128K
  __bf16* vout = (__bf16*)(ws + 589824);     // 64MB value bf16
  float*  pix  = (float*)(ws + 67698688ull); // 20MB pixel coords
  float*  attw = (float*)(ws + 88178688ull); // 10MB attention weights
  __bf16* samp = (__bf16*)(ws + 98418688ull);// 10MB sampled bf16

  // weight repack (L2-resident after this)
  k_wconv<<<256, 256, 0, stream>>>(W_q,    Wqf,  256, 256);
  k_wconv<<<256, 256, 0, stream>>>(W_off,  Woff, 256, 256);
  k_wconv<<<128, 256, 0, stream>>>(W_attn, Watf, 256, 128);
  k_wconv<<<256, 256, 0, stream>>>(W_val,  Wvf,  256, 256);
  k_wconv<<<256, 256, 0, stream>>>(W_out,  Wof,  256, 256);

  k_value <<<VROWS / 32 / 8, 256, 0, stream>>>(fmaps, Wvf, b_val, vout);
  k_query <<<(NQ / 16 + 3) / 4, 128, 0, stream>>>(query, qpos, gamma, beta,
                                                  Wqf, Woff, Watf, b_off, b_attn,
                                                  shapes, pix, attw);
  k_sample<<<(NQ + 7) / 8, 256, 0, stream>>>(pix, attw, vout, qbo, shapes, samp);
  k_out   <<<(NQ / 32 + 7) / 8, 256, 0, stream>>>(samp, Wof, query, out);
}
